// CustomModel_52338471469275
// MI455X (gfx1250) — matrix operand order repaired and checked by reading, not compiled, not run on hardware
//
#include <hip/hip_runtime.h>

// ---------------------------------------------------------------------------
// MI455X (gfx1250) fused kernel for:
//   out = relu(X @ (S*THETA)^T + b1) @ W + b2
// bf16 WMMA (V_WMMA_F32_16X16X32_BF16) with f32 accumulation,
// async global->LDS copies (ASYNCcnt), double-buffered LDS.
// ---------------------------------------------------------------------------

typedef __attribute__((ext_vector_type(16))) __bf16 v16bf;
typedef __attribute__((ext_vector_type(8)))  __bf16 v8bf;
typedef __attribute__((ext_vector_type(4)))  __bf16 v4bf;
typedef __attribute__((ext_vector_type(8)))  float  v8f;

#define BATCH 2048
#define ODIM  1024
#define IDIM  8192

#define BM 128          // block tile M
#define BN 128          // block tile N
#define BK 32           // K step (bf16 WMMA K)
#define LDSS 40         // LDS row stride in bf16 elems (80 B: 16B-aligned, bank-spread)

// workspace layout (bytes)
#define WS_W1B  (0ull)                    // bf16 [ODIM][IDIM]  (S*THETA), 16 MiB
#define WS_XB   (16ull << 20)             // bf16 [BATCH][IDIM], 32 MiB
#define WS_WTB  (48ull << 20)             // bf16 [ODIM][ODIM]  weight^T, 2 MiB
#define WS_FB   (50ull << 20)             // bf16 [BATCH][ODIM] relu(logits), 4 MiB

// ---- async global->LDS 16B copy (per-lane), tracked by ASYNCcnt -----------
__device__ __forceinline__ unsigned lds_lo32(const void* p) {
    // generic shared pointer: addr[31:0] is the LDS byte offset (ISA 10.2)
    return (unsigned)(unsigned long long)p;
}
__device__ __forceinline__ void async_cp16(unsigned lds_off, const void* gsrc) {
#if defined(__HIP_DEVICE_COMPILE__)
    asm volatile("global_load_async_to_lds_b128 %0, %1, off"
                 :: "v"(lds_off), "v"((unsigned long long)gsrc)
                 : "memory");
#else
    (void)lds_off; (void)gsrc;
#endif
}
__device__ __forceinline__ void wait_async0() {
#if defined(__HIP_DEVICE_COMPILE__)
    asm volatile("s_wait_asynccnt 0" ::: "memory");
#endif
}

// Build a v16bf fragment from two 16-byte LDS loads.
__device__ __forceinline__ v16bf load_frag(const __bf16* rowp, int o1, int o2) {
    v8bf lo = *(const v8bf*)(rowp + o1);
    v8bf hi = *(const v8bf*)(rowp + o2);
    return __builtin_shufflevector(lo, hi, 0,1,2,3,4,5,6,7,8,9,10,11,12,13,14,15);
}

// ---------------------------------------------------------------------------
// Prep kernels
// ---------------------------------------------------------------------------
__global__ __launch_bounds__(256) void k_mask_cvt(const float* __restrict__ S,
                                                  const float* __restrict__ T,
                                                  __bf16* __restrict__ W) {
    long i = ((long)blockIdx.x * 256 + threadIdx.x) * 4;
    float4 s = *(const float4*)(S + i);
    float4 t = *(const float4*)(T + i);
    v4bf w;
    w[0] = (__bf16)(s.x * t.x);
    w[1] = (__bf16)(s.y * t.y);
    w[2] = (__bf16)(s.z * t.z);
    w[3] = (__bf16)(s.w * t.w);
    *(v4bf*)(W + i) = w;
}

__global__ __launch_bounds__(256) void k_cvt(const float* __restrict__ X,
                                             __bf16* __restrict__ Y) {
    long i = ((long)blockIdx.x * 256 + threadIdx.x) * 4;
    float4 x = *(const float4*)(X + i);
    v4bf y;
    y[0] = (__bf16)x.x; y[1] = (__bf16)x.y; y[2] = (__bf16)x.z; y[3] = (__bf16)x.w;
    *(v4bf*)(Y + i) = y;
}

// Wt[o][j] = bf16(weight[j][o]) via 32x32 LDS tile
__global__ __launch_bounds__(256) void k_wt_cvt(const float* __restrict__ W,
                                                __bf16* __restrict__ Wt) {
    __shared__ float t[32][33];
    const int bx = blockIdx.x;   // o tile
    const int by = blockIdx.y;   // j tile
    const int tx = threadIdx.x;  // 0..31
    const int ty = threadIdx.y;  // 0..7
    #pragma unroll
    for (int p = 0; p < 4; ++p) {
        int r = ty + 8 * p;
        t[r][tx] = W[(long)(by * 32 + r) * ODIM + bx * 32 + tx];
    }
    __syncthreads();
    #pragma unroll
    for (int p = 0; p < 4; ++p) {
        int r = ty + 8 * p;
        Wt[(long)(bx * 32 + r) * ODIM + by * 32 + tx] = (__bf16)t[tx][r];
    }
}

// ---------------------------------------------------------------------------
// GEMM: C[M][N] = A[M][K](bf16) x Bt[N][K](bf16)^T + bias, optional ReLU.
// RELU=true  -> bf16 output; RELU=false -> f32 output.
// Block: 256 thr (8 waves), tile 128x128; wave tile 32x64 (2x4 frags).
// ---------------------------------------------------------------------------
__device__ __forceinline__ void copy_tile(__bf16 (*dA)[LDSS], __bf16 (*dB)[LDSS],
                                          const __bf16* __restrict__ A,
                                          const __bf16* __restrict__ Bt,
                                          int aRowBase, int bRowBase,
                                          int K, int k0, int tid) {
    #pragma unroll
    for (int p = 0; p < 2; ++p) {
        int c   = tid + p * 256;       // 0..511
        int row = c >> 2;              // 0..127
        int cc  = (c & 3) * 8;         // 0,8,16,24 (bf16 elems; 16B chunks)
        async_cp16(lds_lo32(&dA[row][cc]), A  + (long)(aRowBase + row) * K + k0 + cc);
        async_cp16(lds_lo32(&dB[row][cc]), Bt + (long)(bRowBase + row) * K + k0 + cc);
    }
}

template <bool RELU>
__global__ __launch_bounds__(256) void k_gemm(const __bf16* __restrict__ A,
                                              const __bf16* __restrict__ Bt,
                                              const float* __restrict__ bias,
                                              void* __restrict__ outp,
                                              int N, int K) {
    __shared__ __align__(16) __bf16 sA[2][BM][LDSS];
    __shared__ __align__(16) __bf16 sB[2][BN][LDSS];

    const int tid   = threadIdx.x;
    const int lane  = tid & 31;
    const int wid   = tid >> 5;
    const int waveM = wid & 3;        // 4 waves along M (32 rows each)
    const int waveN = wid >> 2;       // 2 waves along N (64 cols each)
    const int halfK = lane >> 4;      // 0/1: lane group selects K half
    const int l16   = lane & 15;

    const int aRowBase = blockIdx.y * BM;
    const int bRowBase = blockIdx.x * BN;

    v8f acc[2][4];
    #pragma unroll
    for (int m = 0; m < 2; ++m)
        #pragma unroll
        for (int n = 0; n < 4; ++n)
            #pragma unroll
            for (int r = 0; r < 8; ++r) acc[m][n][r] = 0.0f;

    const int nk = K / BK;
    copy_tile(sA[0], sB[0], A, Bt, aRowBase, bRowBase, K, 0, tid);

    for (int kt = 0; kt < nk; ++kt) {
        wait_async0();          // this wave's async tile writes are in LDS
        __syncthreads();        // everyone's writes visible; prev compute done

        if (kt + 1 < nk)
            copy_tile(sA[(kt + 1) & 1], sB[(kt + 1) & 1], A, Bt,
                      aRowBase, bRowBase, K, (kt + 1) * BK, tid);

        const __bf16 (*cA)[LDSS] = sA[kt & 1];
        const __bf16 (*cB)[LDSS] = sB[kt & 1];

        // A frag (16x32 bf16): lane<16 holds K {0..7,16..23}; lane>=16 {8..15,24..31}
        v16bf af[2];
        #pragma unroll
        for (int mf = 0; mf < 2; ++mf)
            af[mf] = load_frag(&cA[waveM * 32 + mf * 16 + l16][0],
                               halfK * 8, halfK * 8 + 16);
        // B frag from n-major LDS: lane<16 holds K {0..7,16..23}; lane>=16 {8..15,24..31}
        v16bf bfm[4];
        #pragma unroll
        for (int nf = 0; nf < 4; ++nf)
            bfm[nf] = load_frag(&cB[waveN * 64 + nf * 16 + l16][0],
                                halfK * 8, halfK * 8 + 16);

        #pragma unroll
        for (int mf = 0; mf < 2; ++mf)
            #pragma unroll
            for (int nf = 0; nf < 4; ++nf)
                acc[mf][nf] = __builtin_amdgcn_wmma_f32_16x16x32_bf16(
                    false, af[mf], false, bfm[nf], (short)0, acc[mf][nf],
                    false, false);
    }

    // Epilogue: C 16x16 layout -> col = l16, row = r + 8*halfK (ISA 7.12.2)
    const int rowBase = aRowBase + waveM * 32 + halfK * 8;
    const int colBase = bRowBase + waveN * 64 + l16;
    #pragma unroll
    for (int mf = 0; mf < 2; ++mf) {
        #pragma unroll
        for (int nf = 0; nf < 4; ++nf) {
            const int c = colBase + nf * 16;
            const float bv = bias[c];
            #pragma unroll
            for (int r = 0; r < 8; ++r) {
                float v = acc[mf][nf][r] + bv;
                const long idx = (long)(rowBase + mf * 16 + r) * N + c;
                if (RELU) {
                    v = v > 0.0f ? v : 0.0f;
                    ((__bf16*)outp)[idx] = (__bf16)v;
                } else {
                    ((float*)outp)[idx] = v;
                }
            }
        }
    }
}

// ---------------------------------------------------------------------------
extern "C" void kernel_launch(void* const* d_in, const int* in_sizes, int n_in,
                              void* d_out, int out_size, void* d_ws, size_t ws_size,
                              hipStream_t stream) {
    (void)in_sizes; (void)n_in; (void)out_size; (void)ws_size;
    const float* input  = (const float*)d_in[0];
    const float* S      = (const float*)d_in[1];
    const float* THETA  = (const float*)d_in[2];
    const float* bias   = (const float*)d_in[3];
    const float* weight = (const float*)d_in[4];
    const float* bias2  = (const float*)d_in[5];

    char* ws = (char*)d_ws;
    __bf16* W1b = (__bf16*)(ws + WS_W1B);   // [ODIM][IDIM] == Bt for GEMM1
    __bf16* Xb  = (__bf16*)(ws + WS_XB);    // [BATCH][IDIM]
    __bf16* Wtb = (__bf16*)(ws + WS_WTB);   // [ODIM][ODIM] == Bt for GEMM2
    __bf16* Fb  = (__bf16*)(ws + WS_FB);    // [BATCH][ODIM]

    // prep: masked weights, input, transposed second-layer weights -> bf16
    k_mask_cvt<<<(ODIM * (long)IDIM) / (256 * 4), 256, 0, stream>>>(S, THETA, W1b);
    k_cvt<<<(BATCH * (long)IDIM) / (256 * 4), 256, 0, stream>>>(input, Xb);
    k_wt_cvt<<<dim3(ODIM / 32, ODIM / 32), dim3(32, 8), 0, stream>>>(weight, Wtb);

    // GEMM1 + bias + ReLU -> bf16 intermediate
    k_gemm<true><<<dim3(ODIM / BN, BATCH / BM), 256, 0, stream>>>(
        Xb, W1b, bias, (void*)Fb, ODIM, IDIM);

    // GEMM2 + bias2 -> f32 output
    k_gemm<false><<<dim3(ODIM / BN, BATCH / BM), 256, 0, stream>>>(
        Fb, Wtb, bias2, d_out, ODIM, ODIM);
}
